// BodyFaceEmotionClassifier_2070174237264
// MI455X (gfx1250) — matrix operand order repaired and checked by reading, not compiled, not run on hardware
//
#include <hip/hip_runtime.h>
#include <stdint.h>

// ---------------------------------------------------------------------------
// MI455X (gfx1250) implementation.
// HBM-bound problem (~211MB in / 23.3 TB/s ≈ 9us floor, ~halved by length
// early-exit); the 18-GFLOP GEMM runs on v_wmma_f32_16x16x32_bf16.
// Wave tile = 64x64 (4 M-tiles x 4 N-tiles): every A/B fragment is reused 4x,
// so fragment loads amortize over 16 back-to-back WMMAs per k-step.
// B fragments stream straight from global (80KB packed W1 stays in WGP$/L2);
// LDS holds only the 128x160 bf16 A tile (+pool) = 44KB, safe in CU-mode.
// ---------------------------------------------------------------------------

typedef __attribute__((ext_vector_type(16))) __bf16      v16bf;
typedef __attribute__((ext_vector_type(8)))  float       v8f;
typedef __attribute__((ext_vector_type(4)))  unsigned int u32x4;

namespace {
constexpr int kB  = 64;     // batch
constexpr int kT  = 4096;   // timesteps
constexpr int kKP = 67;     // keypoints
constexpr int kD  = 256;    // hidden
constexpr int kC  = 7;      // classes
constexpr int kKSteps = 5;  // ceil(134/32): K padded 134 -> 160

constexpr int kRows    = 128;                 // T-rows per block (2 row-groups x 64)
constexpr int kAStride = 168;                 // bf16 per A row in LDS (16B-aligned, bank-friendly)
constexpr int kABytes  = kRows * kAStride * 2;            // 43008
constexpr int kBPackBytes = kKSteps * 16 * 32 * 32;       // 5*16 frags * 32 lanes * 32B = 81920
constexpr int kSmemGemm   = kABytes + kD * 4;             // 44032 < 64KB
}

union Frag { v16bf v; u32x4 q[2]; };

__device__ __forceinline__ unsigned short f32_to_bf16_rne(float f) {
  unsigned u = __float_as_uint(f);
  unsigned r = u + 0x7FFFu + ((u >> 16) & 1u);   // round-to-nearest-even
  return (unsigned short)(r >> 16);
}

// ---------------------------------------------------------------------------
// Repack W1 (f32 [134][256]) into WMMA-B fragment-linear bf16 layout:
// frag (s,n): lane L holds N = 16n + (L&15), K = 32s + 16*(L>>4) + e, e=0..15.
// K >= 134 is zero padding. Output: 80KB in d_ws; each fragment is one
// contiguous, fully-coalesced 1KB burst at GEMM time.
// ---------------------------------------------------------------------------
__global__ void pack_w1_bf16(const float* __restrict__ W1,
                             unsigned short* __restrict__ Bp) {
  int i = blockIdx.x * blockDim.x + threadIdx.x;      // 0..2559
  if (i >= kKSteps * 16 * 32) return;
  int lane = i & 31;
  int frag = i >> 5;                                  // s*16 + n
  int s = frag >> 4, n = frag & 15;
  int h = lane >> 4;
  int col = n * 16 + (lane & 15);
  unsigned short* dst = Bp + (size_t)(frag * 32 + lane) * 16;
#pragma unroll
  for (int e = 0; e < 16; ++e) {
    int k = 32 * s + 16 * h + e;
    float v = (k < 134) ? W1[k * kD + col] : 0.0f;
    dst[e] = f32_to_bf16_rne(v);
  }
}

__global__ void zero_pooled(int* __restrict__ p) {
  p[blockIdx.x * 256 + threadIdx.x] = 0;              // 0 bits == 0.0f (ReLU output >= 0)
}

// ---------------------------------------------------------------------------
// Main fused kernel: gate+convert -> bf16 WMMA GEMM -> bias+ReLU -> masked
// max-pool reduced via LDS then one global atomicMax per feature.
// Block = (b, 128-row T chunk), 256 threads = 8 wave32s arranged as a
// 2 (row-group of 64) x 4 (col-group of 64) grid. Each wave: acc[4][4] v8f.
// ---------------------------------------------------------------------------
__global__ __launch_bounds__(256) void gemm_pool(
    const float* __restrict__ body, const float* __restrict__ hr,
    const float* __restrict__ hl, const int* __restrict__ length,
    const float* __restrict__ b1, const unsigned short* __restrict__ Bp,
    int* __restrict__ pooledBits) {
  extern __shared__ char smem[];
  unsigned short* Ash = (unsigned short*)smem;        // [128][168] bf16
  int* poolsh         = (int*)(smem + kABytes);       // [256]

  const int b  = blockIdx.y;
  const int t0 = blockIdx.x * kRows;
  const int len = length[b];
  if (t0 >= len) return;        // uniform early exit: skip loads AND flops

  const int tid = threadIdx.x;

  poolsh[tid] = 0;
  // zero the K padding [134,160)
  for (int i = tid; i < kRows * 26; i += 256) {
    int r = i / 26, k = 134 + (i - r * 26);
    Ash[r * kAStride + k] = 0;
  }
  // gate coords by confidence, convert to bf16, stage A tile
  for (int i = tid; i < kRows * kKP; i += 256) {
    int r = i / kKP, kp = i - r * kKP;
    int t = t0 + r;
    const float* src;
    if (kp < 25)      src = body + ((b * kT + t) * 75 + 3 * kp);
    else if (kp < 46) src = hr   + ((b * kT + t) * 63 + 3 * (kp - 25));
    else              src = hl   + ((b * kT + t) * 63 + 3 * (kp - 46));
    float x = src[0], y = src[1], c = src[2];
    float g = (c > 0.1f) ? 1.0f : 0.0f;
    unsigned short* dst = &Ash[r * kAStride + 2 * kp];
    dst[0] = f32_to_bf16_rne(x * g);
    dst[1] = f32_to_bf16_rne(y * g);
  }
  __syncthreads();

  const int wave = tid >> 5;
  const int lane = tid & 31;
  const int h    = lane >> 4;
  const int rg   = wave >> 2;                 // row group: 0..1   (64 rows)
  const int cg   = wave & 3;                  // col group: 0..3   (64 cols)
  const int rowbase = rg * 64 + (lane & 15);  // A row for m-tile 0

  const v8f vzero = {};
  v8f acc[4][4];
#pragma unroll
  for (int mi = 0; mi < 4; ++mi)
#pragma unroll
    for (int ni = 0; ni < 4; ++ni) acc[mi][ni] = vzero;

#pragma unroll
  for (int s = 0; s < kKSteps; ++s) {
    // A fragments: lane = row m, K contiguous
    // [32s+16h, +16) (matches packed B)
    Frag a[4];
#pragma unroll
    for (int mi = 0; mi < 4; ++mi) {
      const unsigned short* arow = &Ash[(rowbase + mi * 16) * kAStride + 32 * s];
      a[mi].q[0] = *(const u32x4*)(arow + 16 * h);
      a[mi].q[1] = *(const u32x4*)(arow + 16 * h + 8);
    }
    // B fragments straight from global (WGP$/L2-resident, 1KB coalesced each)
    Frag bb[4];
#pragma unroll
    for (int ni = 0; ni < 4; ++ni) {
      const u32x4* bp = (const u32x4*)Bp +
          (size_t)((s * 16 + cg * 4 + ni) * 32 + lane) * 2;
      bb[ni].q[0] = bp[0];
      bb[ni].q[1] = bp[1];
    }
    // 16 back-to-back WMMAs, every fragment reused 4x
#pragma unroll
    for (int ni = 0; ni < 4; ++ni)
#pragma unroll
      for (int mi = 0; mi < 4; ++mi)
        acc[mi][ni] = __builtin_amdgcn_wmma_f32_16x16x32_bf16(
            false, a[mi].v, false, bb[ni].v, (short)0, acc[mi][ni], false, false);
  }

  // epilogue: +bias, ReLU, masked max over each lane's 8 rows (M = r + 8h),
  // then LDS int-max (exact for non-negative floats) and one global atomic.
#pragma unroll
  for (int ni = 0; ni < 4; ++ni) {
    int ng = cg * 64 + ni * 16 + (lane & 15);
    float bias = b1[ng];
#pragma unroll
    for (int mi = 0; mi < 4; ++mi) {
      float vmax = -1.0f;
#pragma unroll
      for (int r = 0; r < 8; ++r) {
        int t = t0 + rg * 64 + mi * 16 + r + 8 * h;
        float v = fmaxf(acc[mi][ni][r] + bias, 0.0f);
        if (t < len) vmax = fmaxf(vmax, v);
      }
      if (vmax >= 0.0f) atomicMax(&poolsh[ng], __float_as_int(vmax));
    }
  }
  __syncthreads();
  atomicMax(&pooledBits[b * kD + tid], poolsh[tid]);
}

// ---------------------------------------------------------------------------
// BatchNorm (batch stats, ddof=0) + tiny 64x7 classifier. One block.
// ---------------------------------------------------------------------------
__global__ void finalize_bn_fc(const int* __restrict__ pooledBits,
                               const float* __restrict__ gamma,
                               const float* __restrict__ beta,
                               const float* __restrict__ Wc,
                               const float* __restrict__ bc,
                               float* __restrict__ out) {
  extern __shared__ float bnsh[];               // [64][256]
  int tid = threadIdx.x;
  if (tid < kD) {
    int d = tid;
    float sum = 0.f, sumsq = 0.f;
    for (int bb = 0; bb < kB; ++bb) {
      float v = __int_as_float(pooledBits[bb * kD + d]);
      bnsh[bb * kD + d] = v;
      sum += v;
      sumsq += v * v;
    }
    float mean = sum * (1.0f / kB);
    float var  = sumsq * (1.0f / kB) - mean * mean;
    float inv  = rsqrtf(var + 1e-5f);
    float g    = gamma[d] * inv;
    float be   = beta[d] - mean * g;
    for (int bb = 0; bb < kB; ++bb)
      bnsh[bb * kD + d] = bnsh[bb * kD + d] * g + be;
  }
  __syncthreads();
  if (tid < kB * kC) {
    int bb = tid / kC, c = tid - bb * kC;
    float a = bc[c];
    for (int dd = 0; dd < kD; ++dd) a += bnsh[bb * kD + dd] * Wc[dd * kC + c];
    out[bb * kC + c] = a;
  }
}

extern "C" void kernel_launch(void* const* d_in, const int* in_sizes, int n_in,
                              void* d_out, int out_size, void* d_ws, size_t ws_size,
                              hipStream_t stream) {
  (void)in_sizes; (void)n_in; (void)out_size; (void)ws_size;
  const float* body  = (const float*)d_in[0];
  const float* hr    = (const float*)d_in[1];
  const float* hl    = (const float*)d_in[2];
  const int*   len   = (const int*)d_in[3];
  const float* W1    = (const float*)d_in[4];
  const float* b1    = (const float*)d_in[5];
  const float* gamma = (const float*)d_in[6];
  const float* beta  = (const float*)d_in[7];
  const float* Wc    = (const float*)d_in[8];
  const float* bc    = (const float*)d_in[9];

  unsigned short* Bp = (unsigned short*)d_ws;              // 80KB packed W1
  int* pooled = (int*)((char*)d_ws + kBPackBytes);         // 64KB pooled bits

  pack_w1_bf16<<<10, 256, 0, stream>>>(W1, Bp);
  zero_pooled<<<kB, 256, 0, stream>>>(pooled);

  dim3 grid(kT / kRows, kB);
  gemm_pool<<<grid, 256, kSmemGemm, stream>>>(body, hr, hl, len, b1, Bp, pooled);

  finalize_bn_fc<<<1, 512, kB * kD * (int)sizeof(float), stream>>>(
      pooled, gamma, beta, Wc, bc, (float*)d_out);
}
